// MHA_69767448756295
// MI455X (gfx1250) — compile-verified, matrix operand order checked by reading, not run on hardware
//
#include <hip/hip_runtime.h>

// ---------------------------------------------------------------------------
// Problem constants (from reference): B=2, S=2048, E=2048, H=16, HKV=4, D=128
// qkv_dim = D*(H+2*HKV) = 3072
// ---------------------------------------------------------------------------
#define BB   2
#define SS   2048
#define EE   2048
#define HH   16
#define HKV  4
#define DD   128
#define QKVD 3072
#define QSCALE 0.08838834764831845f  /* 1/sqrt(128) */

typedef __attribute__((ext_vector_type(16))) __bf16 v16bf;
typedef __attribute__((ext_vector_type(8)))  __bf16 v8bf;
typedef __attribute__((ext_vector_type(8)))  float  v8f;

// ---------------------------------------------------------------------------
// WMMA helper: D = A(16x32 bf16) * B(32x16 bf16) + C(16x16 f32)
// ---------------------------------------------------------------------------
__device__ __forceinline__ v8f wmma_bf16(v16bf a, v16bf b, v8f c) {
    return __builtin_amdgcn_wmma_f32_16x16x32_bf16(
        /*neg_a=*/false, a, /*neg_b=*/false, b,
        /*c_mod=*/(short)0, c, /*reuse_a=*/false, /*reuse_b=*/false);
}

// Load a 16x32 fragment (A layout; identical for NT-GEMM B operand) from a
// row-major [rows, ld] bf16 matrix (global or LDS). Per the 16-bit A layout:
//   lanes 0-15  : M = lane,    K in {k0+0..7,  k0+16..23}
//   lanes 16-31 : M = lane-16, K in {k0+8..15, k0+24..31}
__device__ __forceinline__ v16bf load_frag(const __bf16* base,
                                           int ld, int row0, int k0) {
    const int lane = threadIdx.x & 31;
    const int r    = row0 + (lane & 15);
    const int kb   = k0 + ((lane >> 4) << 3);
    const __bf16* p = base + (long)r * ld + kb;
    v16bf f;
    *(v8bf*)&f       = *(const v8bf*)(p);
    *((v8bf*)&f + 1) = *(const v8bf*)(p + 16);
    return f;
}

// gfx1250 async copy global -> LDS (16 bytes per lane), tracked by ASYNCcnt.
__device__ __forceinline__ void async_copy_b128(unsigned lds_addr,
                                                const void* gaddr) {
    asm volatile("global_load_async_to_lds_b128 %0, %1, off"
                 :: "v"(lds_addr), "v"(gaddr) : "memory");
}
__device__ __forceinline__ unsigned lds_off(const void* p) {
    // generic pointers to LDS carry the offset in the low 32 bits
    return (unsigned)(size_t)p;
}

// ---------------------------------------------------------------------------
// Kernel: fp32 -> bf16 conversion
// ---------------------------------------------------------------------------
__global__ void k_f32_to_bf16(const float* __restrict__ in,
                              __bf16* __restrict__ out, long n) {
    long i = (long)blockIdx.x * blockDim.x + threadIdx.x;
    if (i < n) out[i] = (__bf16)in[i];
}

// ---------------------------------------------------------------------------
// Kernel: NT GEMM  C[M,N] = A[M,K] * B[N,K]^T + bias[N]   (bf16 in, f32 out)
// grid: (N/128, M/128), block: 256 (8 waves, 4(M) x 2(N)).
// Double-buffered async staging of 128x32 A/B tiles into LDS; each wave
// computes a 32x64 sub-tile (2x4 WMMA accumulators) from LDS fragments.
// ---------------------------------------------------------------------------
#define MT 128
#define NT 128
#define KT 32

__global__ __launch_bounds__(256)
void k_gemm_nt_bf16(const __bf16* __restrict__ A, const __bf16* __restrict__ Bm,
                    const float* __restrict__ bias, float* __restrict__ C,
                    int M, int N, int K) {
    __shared__ __attribute__((aligned(16))) __bf16 ta[2][MT * KT];
    __shared__ __attribute__((aligned(16))) __bf16 tb[2][NT * KT];

    const int tid  = threadIdx.x;
    const int wave = tid >> 5;
    const int lane = tid & 31;
    const int wm   = wave & 3;            // M sub-block (x32)
    const int wn   = wave >> 2;           // N sub-block (x64)
    const int m_blk = blockIdx.y * MT;
    const int n_blk = blockIdx.x * NT;

    // Stage one 128x32 K-chunk of A and B into LDS buffer `buf`.
    // 512 16-byte chunks per tile, 2 per thread per matrix -> 4 async
    // instructions per wave per chunk (ASYNCcnt += 4).
    auto issue = [&](int buf, int k0) {
        for (int c = tid; c < (MT * KT) / 8; c += 256) {
            int row  = c >> 2;
            int quad = (c & 3) * 8;               // 8 bf16 = 16 bytes
            async_copy_b128(lds_off(&ta[buf][row * KT + quad]),
                            A  + (long)(m_blk + row) * K + k0 + quad);
            async_copy_b128(lds_off(&tb[buf][row * KT + quad]),
                            Bm + (long)(n_blk + row) * K + k0 + quad);
        }
    };

    v8f acc[2][4] = {};
    issue(0, 0);
    const int nk = K / KT;
    for (int it = 0; it < nk; ++it) {
        const int  cur  = it & 1;
        const bool more = (it + 1) < nk;
        if (more) {
            issue(cur ^ 1, (it + 1) * KT);
            // tile `cur` (4 older async ops) complete when <=4 outstanding
            asm volatile("s_wait_asynccnt 0x4" ::: "memory");
        } else {
            asm volatile("s_wait_asynccnt 0x0" ::: "memory");
        }
        __syncthreads();

        v16bf af[2];
#pragma unroll
        for (int i = 0; i < 2; ++i)
            af[i] = load_frag(&ta[cur][0], KT, wm * 32 + i * 16, 0);
#pragma unroll
        for (int j = 0; j < 4; ++j) {
            v16bf bfr = load_frag(&tb[cur][0], KT, wn * 64 + j * 16, 0);
#pragma unroll
            for (int i = 0; i < 2; ++i)
                acc[i][j] = wmma_bf16(af[i], bfr, acc[i][j]);
        }
        __syncthreads();   // all waves done reading `cur` before overwrite
    }

    // C fragment layout: VGPR i -> row i + 8*(lane>=16), lane&15 -> col
    const int half = lane >> 4;
    const int col  = lane & 15;
#pragma unroll
    for (int i2 = 0; i2 < 2; ++i2)
#pragma unroll
        for (int j = 0; j < 4; ++j)
#pragma unroll
            for (int i = 0; i < 8; ++i) {
                int r = m_blk + wm * 32 + i2 * 16 + i + 8 * half;
                int c = n_blk + wn * 64 + j * 16 + col;
                C[(long)r * N + c] = acc[i2][j][i] + bias[c];
            }
}

// ---------------------------------------------------------------------------
// Kernel: depthwise causal conv (kernel width 4, left pad 3) + conv bias
// ---------------------------------------------------------------------------
__global__ void k_dwconv(const float* __restrict__ qkv,
                         const float* __restrict__ cw,
                         const float* __restrict__ cb,
                         float* __restrict__ out) {
    long idx = (long)blockIdx.x * blockDim.x + threadIdx.x;
    const long total = (long)BB * SS * QKVD;
    if (idx >= total) return;
    int c = (int)(idx % QKVD);
    long r = idx / QKVD;
    int s = (int)(r % SS);
    int b = (int)(r / SS);
    float acc = cb[c];
#pragma unroll
    for (int j = 0; j < 4; ++j) {
        int sp = s - 3 + j;
        if (sp >= 0)
            acc += qkv[((long)b * SS + sp) * QKVD + c] * cw[c * 4 + j];
    }
    out[idx] = acc;
}

// ---------------------------------------------------------------------------
// Kernel: RoPE + split + layout for attention.
//   q -> bf16 [B,H,S,D]   (rotary applied, pre-scaled by 1/sqrt(D))
//   k -> bf16 [B,HKV,S,D] (rotary applied)
//   v -> bf16 [B,HKV,D,S] (transposed so PV GEMM is NT)
// ---------------------------------------------------------------------------
__global__ void k_rope_split(const float* __restrict__ qkv2,
                             __bf16* __restrict__ qb,
                             __bf16* __restrict__ kb,
                             __bf16* __restrict__ vt) {
    long idx = (long)blockIdx.x * blockDim.x + threadIdx.x;
    const long nq = (long)BB * SS * HH  * 64;
    const long nk = (long)BB * SS * HKV * 64;
    const long nv = (long)BB * SS * HKV * DD;
    if (idx < nq) {
        int  i = (int)(idx % 64);  long r = idx / 64;
        int  h = (int)(r % HH);    r /= HH;
        int  s = (int)(r % SS);    int b = (int)(r / SS);
        const float* row = qkv2 + ((long)b * SS + s) * QKVD + h * DD;
        float x1 = row[i], x2 = row[i + 64];
        float ang = (float)s * __powf(10000.0f, -(2.0f * i) / 128.0f);
        float sn, cs; __sincosf(ang, &sn, &cs);
        __bf16* o = qb + (((long)b * HH + h) * SS + s) * DD + i;
        o[0]  = (__bf16)((x1 * cs - x2 * sn) * QSCALE);
        o[64] = (__bf16)((x2 * cs + x1 * sn) * QSCALE);
    } else if (idx < nq + nk) {
        long j = idx - nq;
        int  i = (int)(j % 64);  long r = j / 64;
        int  hk = (int)(r % HKV); r /= HKV;
        int  s = (int)(r % SS);  int b = (int)(r / SS);
        const float* row = qkv2 + ((long)b * SS + s) * QKVD + HH * DD + hk * DD;
        float x1 = row[i], x2 = row[i + 64];
        float ang = (float)s * __powf(10000.0f, -(2.0f * i) / 128.0f);
        float sn, cs; __sincosf(ang, &sn, &cs);
        __bf16* o = kb + (((long)b * HKV + hk) * SS + s) * DD + i;
        o[0]  = (__bf16)(x1 * cs - x2 * sn);
        o[64] = (__bf16)(x2 * cs + x1 * sn);
    } else if (idx < nq + nk + nv) {
        long j = idx - nq - nk;
        int  d = (int)(j % DD);  long r = j / DD;
        int  hk = (int)(r % HKV); r /= HKV;
        int  s = (int)(r % SS);  int b = (int)(r / SS);
        float v = qkv2[((long)b * SS + s) * QKVD + HH * DD + HKV * DD + hk * DD + d];
        vt[(((long)b * HKV + hk) * DD + d) * SS + s] = (__bf16)v;
    }
}

// ---------------------------------------------------------------------------
// Kernel: causal GQA flash attention. One wave per 16 q-rows per (b,h).
// Online softmax runs in the WMMA C-fragment layout; the P tile is pushed
// through a 16x32 LDS staging buffer to re-shape it into the A-fragment
// layout for the PV WMMA (K = t dimension).
// ---------------------------------------------------------------------------
__global__ __launch_bounds__(32)
void k_flash_attn(const __bf16* __restrict__ q, const __bf16* __restrict__ k,
                  const __bf16* __restrict__ vt, __bf16* __restrict__ ctx) {
    __shared__ __attribute__((aligned(16))) __bf16 pbuf[16 * 32];

    const int lane = threadIdx.x & 31;
    const int half = lane >> 4;
    const int col  = lane & 15;
    const int s0 = blockIdx.x * 16;
    const int h  = blockIdx.y;
    const int b  = blockIdx.z;
    const int hk = h / (HH / HKV);

    const __bf16* qbase = q  + ((long)b * HH  + h)  * SS * DD;
    const __bf16* kbase = k  + ((long)b * HKV + hk) * SS * DD;
    const __bf16* vbase = vt + ((long)b * HKV + hk) * DD * SS;

    v16bf qf[4];
#pragma unroll
    for (int kc = 0; kc < 4; ++kc) qf[kc] = load_frag(qbase, DD, s0, kc * 32);

    v8f acc[8] = {};
    float m8[8], l8[8];
#pragma unroll
    for (int i = 0; i < 8; ++i) { m8[i] = -3.0e38f; l8[i] = 0.0f; }

    for (int t0 = 0; t0 < s0 + 16; t0 += 32) {
        v8f slo = {}, shi = {};
#pragma unroll
        for (int kc = 0; kc < 4; ++kc) {
            v16bf klo = load_frag(kbase, DD, t0,      kc * 32);
            v16bf khi = load_frag(kbase, DD, t0 + 16, kc * 32);
            slo = wmma_bf16(qf[kc], klo, slo);
            shi = wmma_bf16(qf[kc], khi, shi);
        }

        float sc8[8];
#pragma unroll
        for (int i = 0; i < 8; ++i) {
            const int row = s0 + i + 8 * half;
            float a  = (t0 + col      <= row) ? slo[i] : -3.0e38f;
            float bb = (t0 + 16 + col <= row) ? shi[i] : -3.0e38f;
            float t = fmaxf(a, bb);
#pragma unroll
            for (int off = 8; off >= 1; off >>= 1)
                t = fmaxf(t, __shfl_xor(t, off, 32));
            float mn = fmaxf(m8[i], t);
            float scale = __expf(m8[i] - mn);
            float plo = __expf(a - mn);
            float phi = __expf(bb - mn);
            float rs = plo + phi;
#pragma unroll
            for (int off = 8; off >= 1; off >>= 1)
                rs += __shfl_xor(rs, off, 32);
            l8[i] = l8[i] * scale + rs;
            m8[i] = mn;
            sc8[i] = scale;
            pbuf[(i + 8 * half) * 32 + col]      = (__bf16)plo;
            pbuf[(i + 8 * half) * 32 + 16 + col] = (__bf16)phi;
        }
        __syncthreads();

        v16bf pf;
        *(v8bf*)&pf       = *(const v8bf*)(pbuf + col * 32 + half * 8);
        *((v8bf*)&pf + 1) = *(const v8bf*)(pbuf + col * 32 + 16 + half * 8);

#pragma unroll
        for (int n = 0; n < 8; ++n) {
#pragma unroll
            for (int i = 0; i < 8; ++i) acc[n][i] *= sc8[i];
            v16bf vf = load_frag(vbase, SS, n * 16, t0);
            acc[n] = wmma_bf16(pf, vf, acc[n]);
        }
        __syncthreads();
    }

#pragma unroll
    for (int n = 0; n < 8; ++n)
#pragma unroll
        for (int i = 0; i < 8; ++i) {
            int row = s0 + i + 8 * half;
            float o = acc[n][i] / l8[i];
            ctx[((long)b * SS + row) * (HH * DD) + h * DD + n * 16 + col] = (__bf16)o;
        }
}

// ---------------------------------------------------------------------------
// Host side
// ---------------------------------------------------------------------------
extern "C" void kernel_launch(void* const* d_in, const int* in_sizes, int n_in,
                              void* d_out, int out_size, void* d_ws, size_t ws_size,
                              hipStream_t stream) {
    const float* x      = (const float*)d_in[0];
    const float* w_in   = (const float*)d_in[1];
    const float* b_in   = (const float*)d_in[2];
    const float* conv_w = (const float*)d_in[3];
    const float* conv_b = (const float*)d_in[4];
    const float* w_out  = (const float*)d_in[5];
    const float* b_out  = (const float*)d_in[6];
    float* out = (float*)d_out;

    char* ws = (char*)d_ws;
    size_t off = 0;
    auto alloc = [&](size_t bytes) -> void* {
        void* p = ws + off;
        off += (bytes + 255) & ~(size_t)255;
        return p;
    };
    __bf16* xb   = (__bf16*)alloc((size_t)BB * SS * EE * 2);
    __bf16* wib  = (__bf16*)alloc((size_t)QKVD * EE * 2);
    __bf16* wob  = (__bf16*)alloc((size_t)EE * HH * DD * 2);
    float*  qkv  = (float*) alloc((size_t)BB * SS * QKVD * 4);
    float*  qkv2 = (float*) alloc((size_t)BB * SS * QKVD * 4);
    __bf16* qb   = (__bf16*)alloc((size_t)BB * HH  * SS * DD * 2);
    __bf16* kb   = (__bf16*)alloc((size_t)BB * HKV * SS * DD * 2);
    __bf16* vt   = (__bf16*)alloc((size_t)BB * HKV * DD * SS * 2);
    __bf16* ctxb = (__bf16*)alloc((size_t)BB * SS * HH * DD * 2);

    const long nx  = (long)BB * SS * EE;
    const long nwi = (long)QKVD * EE;
    const long nwo = (long)EE * HH * DD;
    k_f32_to_bf16<<<dim3((nx  + 255) / 256), 256, 0, stream>>>(x,     xb,  nx);
    k_f32_to_bf16<<<dim3((nwi + 255) / 256), 256, 0, stream>>>(w_in,  wib, nwi);
    k_f32_to_bf16<<<dim3((nwo + 255) / 256), 256, 0, stream>>>(w_out, wob, nwo);

    // QKV projection: [4096,2048] x [3072,2048]^T + b_in
    k_gemm_nt_bf16<<<dim3(QKVD / NT, (BB * SS) / MT), 256, 0, stream>>>(
        xb, wib, b_in, qkv, BB * SS, QKVD, EE);

    // depthwise causal conv
    const long nconv = (long)BB * SS * QKVD;
    k_dwconv<<<dim3((nconv + 255) / 256), 256, 0, stream>>>(qkv, conv_w, conv_b, qkv2);

    // RoPE + split + v transpose
    const long nrope = (long)BB * SS * (HH * 64 + HKV * 64 + HKV * DD);
    k_rope_split<<<dim3((nrope + 255) / 256), 256, 0, stream>>>(qkv2, qb, kb, vt);

    // flash attention: grid (S/16, H, B), one wave per block
    k_flash_attn<<<dim3(SS / 16, HH, BB), 32, 0, stream>>>(qb, kb, vt, ctxb);

    // out projection: [4096,2048] x [2048,2048]^T + b_out -> d_out
    k_gemm_nt_bf16<<<dim3(EE / NT, (BB * SS) / MT), 256, 0, stream>>>(
        ctxb, wob, b_out, out, BB * SS, EE, HH * DD);
}
